// RITS_71983651881270
// MI455X (gfx1250) — matrix operand order repaired and checked by reading, not compiled, not run on hardware
//
#include <hip/hip_runtime.h>
#include <hip/hip_bf16.h>
#include <stddef.h>

// Problem constants (match reference)
#define BB 1024
#define SS 256
#define DD 128
#define HH 512
#define KIH 288           // 257 padded up to 9 * 32
#define G4H (4 * HH)      // 2048

typedef __attribute__((ext_vector_type(16))) _Float16 v16h;
typedef __attribute__((ext_vector_type(8)))  _Float16 v8h;
typedef __attribute__((ext_vector_type(8)))  float    v8f;

// ---------------- workspace layout (bytes) ----------------
#define OFF_WIH 0                                   // 2048*288 f16 = 1,179,648
#define OFF_WHH (OFF_WIH + G4H * KIH * 2)           // 2048*512 f16 = 2,097,152
#define OFF_WHR (OFF_WHH + G4H * HH * 2)            // 128*512  f16 =   131,072
#define OFF_WFR (OFF_WHR + DD * HH * 2)             // 128*128  f16 =    32,768
#define OFF_BSUM (OFF_WFR + DD * DD * 2)            // 2048 f32
#define OFF_NUM  (OFF_BSUM + G4H * 4)               // 256 f32
#define OFF_DEN  (OFF_NUM + SS * 4)                 // 256 f32
// total ~3.45 MB

// ---------------- helpers ----------------
static __device__ __forceinline__ v8f splat8(float x) {
    return (v8f){x, x, x, x, x, x, x, x};
}
static __device__ __forceinline__ float sigm(float x) {
    return 1.0f / (1.0f + __expf(-x));
}
static __device__ __forceinline__ float fast_tanh(float x) {
    // tanh(x) = 2*sigmoid(2x) - 1 : one v_exp_f32 instead of a libm call chain
    return 2.0f / (1.0f + __expf(-2.0f * x)) - 1.0f;
}

struct AFragPair { v8h lo; v8h hi; };

// A fragment (16x32 f16, M rows x K): lane<16 -> M=lane, K = k0+{0..7, 16..23}
//                                     lane>=16 -> M=lane-16, K = k0+{8..15, 24..31}
static __device__ __forceinline__ v16h load_a16(const _Float16* base, int lda,
                                                int k0, int lane) {
    int m  = lane & 15;
    int kb = (lane < 16) ? 0 : 8;
    const _Float16* p = base + m * lda + k0 + kb;
    AFragPair u;
    u.lo = *(const v8h*)(p);
    u.hi = *(const v8h*)(p + 16);
    return __builtin_bit_cast(v16h, u);
}
// B fragment (32x16 f16, K x N) from row-major W[N][K] (i.e. W^T is B):
// lane<16 -> K = k0+{0..7, 16..23}, lane>=16 -> K = k0+{8..15, 24..31}, N = n0 + (lane&15)
static __device__ __forceinline__ v16h load_b16(const _Float16* W, int ldb,
                                                int n0, int k0, int lane) {
    int n  = n0 + (lane & 15);
    int kb = (lane < 16) ? 0 : 8;
    const _Float16* p = W + (size_t)n * ldb + k0 + kb;
    AFragPair u;
    u.lo = *(const v8h*)(p);
    u.hi = *(const v8h*)(p + 16);
    return __builtin_bit_cast(v16h, u);
}
static __device__ __forceinline__ v8f wmma_f16(v16h a, v16h b, v8f c) {
    return __builtin_amdgcn_wmma_f32_16x16x32_f16(false, a, false, b,
                                                  (short)0, c, false, false);
}

// ---------------- kernel 0: zero loss accumulators ----------------
__global__ void rits_init(float* num, float* den) {
    int i = threadIdx.x;
    if (i < SS) { num[i] = 0.0f; den[i] = 0.0f; }
}

// ---------------- kernel 1: convert weights fp32 -> fp16 (padded) ----------------
__global__ void rits_convert(const float* Wih, const float* Whh,
                             const float* Whr, const float* Wfr,
                             const float* bih, const float* bhh,
                             _Float16* WihH, _Float16* WhhH,
                             _Float16* WhrH, _Float16* WfrH, float* bsum) {
    const size_t N1 = (size_t)G4H * KIH;
    const size_t N2 = (size_t)G4H * HH;
    const size_t N3 = (size_t)DD * HH;
    const size_t N4 = (size_t)DD * DD;
    const size_t N5 = (size_t)G4H;
    size_t idx = (size_t)blockIdx.x * blockDim.x + threadIdx.x;
    if (idx < N1) {
        size_t n = idx / KIH, k = idx % KIH;
        WihH[idx] = (k < 257) ? (_Float16)Wih[n * 257 + k] : (_Float16)0.0f;
    } else if (idx < N1 + N2) {
        size_t j = idx - N1;
        WhhH[j] = (_Float16)Whh[j];
    } else if (idx < N1 + N2 + N3) {
        size_t j = idx - N1 - N2;
        WhrH[j] = (_Float16)Whr[j];
    } else if (idx < N1 + N2 + N3 + N4) {
        size_t j = idx - N1 - N2 - N3;
        WfrH[j] = (_Float16)Wfr[j];
    } else if (idx < N1 + N2 + N3 + N4 + N5) {
        size_t j = idx - N1 - N2 - N3 - N4;
        bsum[j] = bih[j] + bhh[j];
    }
}

// ---------------- kernel 2: persistent per-batch-tile LSTM ----------------
// grid = B/16 blocks; block = 512 threads (16 waves). Each block runs all S steps
// for its 16 batch rows; waves split the N dimension of every GEMM.
__global__ __launch_bounds__(512, 1)
void rits_main(const float* __restrict__ data, const float* __restrict__ masks,
               const float* __restrict__ deltas,
               const float* __restrict__ Wtd, const float* __restrict__ btd,
               const float* __restrict__ bhr, const float* __restrict__ bfr,
               const _Float16* __restrict__ WihH, const _Float16* __restrict__ WhhH,
               const _Float16* __restrict__ WhrH, const _Float16* __restrict__ WfrH,
               const float* __restrict__ bsum,
               float* __restrict__ out, float* __restrict__ gNum,
               float* __restrict__ gDen) {
    __shared__ _Float16 hbuf[16 * HH];     // h (f16, A-matrix layout source) 16 KB
    __shared__ _Float16 xcb[16 * KIH];     // [x_c | m | d | 0-pad]            9 KB
    __shared__ _Float16 x16[16 * DD];      // x as f16 (A for Wfr GEMM)        4 KB
    __shared__ float    x32[16 * DD];      // x f32                            8 KB
    __shared__ float    m32[16 * DD];      // m f32                            8 KB
    __shared__ float    xh2[16 * DD];      // h_new @ Whr^T                    8 KB
    __shared__ float    xfs[16 * DD];      // x @ Wfr^T                        8 KB
    __shared__ float    dts[16];           // delta_t per row

    const int tid  = threadIdx.x;
    const int lane = tid & 31;
    const int wv   = tid >> 5;             // 0..15
    const int ln   = lane & 15;
    const int moff = (lane < 16) ? 0 : 8;
    const int r0   = blockIdx.x * 16;      // batch tile base

    // per-thread decay coefficients (column j == tid, hoisted out of t loop)
    const float wtdj = Wtd[tid];
    const float btdj = btd[tid];

    // phase-1/6 element mapping: idx = tid + i*512 -> row rb+4i, col cfix
    const int rb   = tid >> 7;             // 0..3
    const int cfix = tid & 127;

    // init: h = 0, zero-pad xc cols 257..287, c = 0 in registers
#pragma unroll
    for (int i = 0; i < 16; ++i) hbuf[tid + i * 512] = (_Float16)0.0f;
    if (tid < 16 * 31) {
        int r = tid / 31, c2 = 257 + tid % 31;
        xcb[r * KIH + c2] = (_Float16)0.0f;
    }
    v8f creg0 = splat8(0.0f), creg1 = splat8(0.0f);
    __syncthreads();

    for (int t = 0; t < SS; ++t) {
        // ---- phase 0: stage delta_t for the 16 rows ----
        if (tid < 16) dts[tid] = deltas[(size_t)(r0 + tid) * SS + t];
        __syncthreads();

        // ---- phase 1: in-place decay of h; stage x, m ----
        {
            _Float16* hb = hbuf + tid;                 // offsets r*512 elements
#pragma unroll 4
            for (int r = 0; r < 16; ++r) {
                float dec = __expf(-fmaxf(dts[r] * wtdj + btdj, 0.0f));
                hb[r * 512] = (_Float16)((float)hb[r * 512] * dec);
            }
            const float* gdp = data  + ((size_t)(r0 + rb) * SS + t) * DD + cfix;
            const float* gmp = masks + ((size_t)(r0 + rb) * SS + t) * DD + cfix;
            float*    x32p = x32 + tid;
            float*    m32p = m32 + tid;
            _Float16* x16p = x16 + tid;
            _Float16* xcmp = xcb + rb * KIH + 128 + cfix;
#pragma unroll
            for (int i = 0; i < 4; ++i) {
                float xv = gdp[(size_t)i * 4 * SS * DD];
                float mv = gmp[(size_t)i * 4 * SS * DD];
                x32p[i * 512] = xv;
                m32p[i * 512] = mv;
                x16p[i * 512] = (_Float16)xv;
                xcmp[i * 4 * KIH] = (_Float16)mv;
            }
            if (tid < 16) xcb[tid * KIH + 256] = (_Float16)dts[tid];
        }
        __syncthreads();

        // ---- phase 2: waves 0..7: x_h = h_dec @ Whr^T, build x_c columns;
        //               waves 8..15: x_f = x @ Wfr^T (independent of h_new) ----
        if (wv < 8) {
            int fc0 = wv * 16;
            v8f acc = splat8(bhr[fc0 + ln]);
#pragma unroll 2
            for (int kc = 0; kc < 16; ++kc) {
                v16h a = load_a16(hbuf, HH, kc * 32, lane);
                v16h b = load_b16(WhrH, HH, fc0, kc * 32, lane);
                acc = wmma_f16(a, b, acc);
            }
            const float* mb = m32 + moff * DD + fc0 + ln;   // +e*DD
            const float* xb = x32 + moff * DD + fc0 + ln;
            _Float16*   xcw = xcb + moff * KIH + fc0 + ln;  // +e*KIH
#pragma unroll
            for (int e = 0; e < 8; ++e) {
                float mv = mb[e * DD], xv = xb[e * DD];
                xcw[e * KIH] = (_Float16)(mv * xv + (1.0f - mv) * acc[e]);
            }
        } else {
            int fc0 = (wv - 8) * 16;
            v8f acc = splat8(bfr[fc0 + ln]);
#pragma unroll 2
            for (int kc = 0; kc < 4; ++kc) {
                v16h a = load_a16(x16, DD, kc * 32, lane);
                v16h b = load_b16(WfrH, DD, fc0, kc * 32, lane);
                acc = wmma_f16(a, b, acc);
            }
            float* ob = xfs + moff * DD + fc0 + ln;
#pragma unroll
            for (int e = 0; e < 8; ++e) ob[e * DD] = acc[e];
        }
        __syncthreads();

        // ---- phase 3: gates + LSTM cell (all 16 waves, 2 hidden tiles each) ----
        v8f hnew[2];
#pragma unroll
        for (int ht = 0; ht < 2; ++ht) {
            int hc0 = wv * 32 + ht * 16;
            v8f acc[4];
#pragma unroll
            for (int g = 0; g < 4; ++g) acc[g] = splat8(bsum[g * HH + hc0 + ln]);
            // inp @ Wih^T  (K = 288, 9 chunks)
#pragma unroll 2
            for (int kc = 0; kc < 9; ++kc) {
                v16h a = load_a16(xcb, KIH, kc * 32, lane);
#pragma unroll
                for (int g = 0; g < 4; ++g) {
                    v16h b = load_b16(WihH, KIH, g * HH + hc0, kc * 32, lane);
                    acc[g] = wmma_f16(a, b, acc[g]);
                }
            }
            // h_dec @ Whh^T  (K = 512, 16 chunks)
#pragma unroll 2
            for (int kc = 0; kc < 16; ++kc) {
                v16h a = load_a16(hbuf, HH, kc * 32, lane);
#pragma unroll
                for (int g = 0; g < 4; ++g) {
                    v16h b = load_b16(WhhH, HH, g * HH + hc0, kc * 32, lane);
                    acc[g] = wmma_f16(a, b, acc[g]);
                }
            }
            // cell update, fully wave-local (i,f,g,o share the element mapping)
            v8f cc = ht ? creg1 : creg0;
            v8f hv;
#pragma unroll
            for (int e = 0; e < 8; ++e) {
                float iv = sigm(acc[0][e]);
                float fv = sigm(acc[1][e]);
                float gv = fast_tanh(acc[2][e]);
                float ov = sigm(acc[3][e]);
                float cn = fv * cc[e] + iv * gv;
                cc[e] = cn;
                hv[e] = ov * fast_tanh(cn);
            }
            if (ht) creg1 = cc; else creg0 = cc;
            hnew[ht] = hv;
        }
        __syncthreads();   // all reads of hbuf done -> safe to overwrite

        // ---- phase 4: publish h_new as f16 A-matrix source ----
        {
            _Float16* hw = hbuf + moff * HH + wv * 32 + ln;   // +e*HH, +ht*16
#pragma unroll
            for (int ht = 0; ht < 2; ++ht)
#pragma unroll
                for (int e = 0; e < 8; ++e)
                    hw[e * HH + ht * 16] = (_Float16)hnew[ht][e];
        }
        __syncthreads();

        // ---- phase 5: x_h2 = h_new @ Whr^T (waves 0..7 only; x_f done in ph2) ----
        if (wv < 8) {
            int fc0 = wv * 16;
            v8f acc = splat8(bhr[fc0 + ln]);
#pragma unroll 2
            for (int kc = 0; kc < 16; ++kc) {
                v16h a = load_a16(hbuf, HH, kc * 32, lane);
                v16h b = load_b16(WhrH, HH, fc0, kc * 32, lane);
                acc = wmma_f16(a, b, acc);
            }
            float* ob = xh2 + moff * DD + fc0 + ln;
#pragma unroll
            for (int e = 0; e < 8; ++e) ob[e * DD] = acc[e];
        }
        __syncthreads();

        // ---- phase 6: x_comb -> out; masked SE loss partials ----
        float nu = 0.0f, de = 0.0f;
        {
            const float* x32p = x32 + tid;
            const float* m32p = m32 + tid;
            const float* h2p  = xh2 + tid;
            const float* ffp  = xfs + tid;
            float* op = out + ((size_t)(r0 + rb) * SS + t) * DD + cfix;
#pragma unroll
            for (int i = 0; i < 4; ++i) {
                float xv = x32p[i * 512], mv = m32p[i * 512];
                float h2 = h2p[i * 512], ff = ffp[i * 512];
                op[(size_t)i * 4 * SS * DD] = mv * xv + (1.0f - mv) * (h2 + ff);
                float dd = xv - h2;
                nu += dd * dd * mv;
                de += mv;
            }
        }
#pragma unroll
        for (int off = 16; off > 0; off >>= 1) {
            nu += __shfl_xor(nu, off, 32);
            de += __shfl_xor(de, off, 32);
        }
        if (lane == 0) {
            atomicAdd(&gNum[t], nu);
            atomicAdd(&gDen[t], de);
        }
        __syncthreads();   // protect x32/m32/xh2/xfs/dts before next step
    }
}

// ---------------- kernel 3: finalize loss ----------------
__global__ void rits_finalize(const float* num, const float* den, float* lossOut) {
    __shared__ float red[SS];
    int t = threadIdx.x;
    red[t] = num[t] / (den[t] + 1e-5f);
    __syncthreads();
    for (int s = SS / 2; s > 0; s >>= 1) {
        if (t < s) red[t] += red[t + s];
        __syncthreads();
    }
    if (t == 0) *lossOut = red[0] / (float)SS;
}

// ---------------- host launcher ----------------
extern "C" void kernel_launch(void* const* d_in, const int* in_sizes, int n_in,
                              void* d_out, int out_size, void* d_ws, size_t ws_size,
                              hipStream_t stream) {
    // setup_inputs() dict order:
    const float* data   = (const float*)d_in[0];
    const float* masks  = (const float*)d_in[1];
    const float* deltas = (const float*)d_in[2];
    const float* Wih    = (const float*)d_in[3];
    const float* Whh    = (const float*)d_in[4];
    const float* bih    = (const float*)d_in[5];
    const float* bhh    = (const float*)d_in[6];
    const float* Wtd    = (const float*)d_in[7];
    const float* btd    = (const float*)d_in[8];
    const float* Whr    = (const float*)d_in[9];
    const float* bhr    = (const float*)d_in[10];
    const float* Wfr    = (const float*)d_in[11];
    const float* bfr    = (const float*)d_in[12];

    char* ws = (char*)d_ws;
    _Float16* WihH = (_Float16*)(ws + OFF_WIH);
    _Float16* WhhH = (_Float16*)(ws + OFF_WHH);
    _Float16* WhrH = (_Float16*)(ws + OFF_WHR);
    _Float16* WfrH = (_Float16*)(ws + OFF_WFR);
    float*    bsum = (float*)(ws + OFF_BSUM);
    float*    gNum = (float*)(ws + OFF_NUM);
    float*    gDen = (float*)(ws + OFF_DEN);

    float* out     = (float*)d_out;                 // imputations [B,S,D]
    float* lossOut = out + (size_t)BB * SS * DD;    // scalar loss appended

    rits_init<<<1, SS, 0, stream>>>(gNum, gDen);

    const size_t convTotal = (size_t)G4H * KIH + (size_t)G4H * HH +
                             (size_t)DD * HH + (size_t)DD * DD + (size_t)G4H;
    int convBlocks = (int)((convTotal + 255) / 256);
    rits_convert<<<convBlocks, 256, 0, stream>>>(Wih, Whh, Whr, Wfr, bih, bhh,
                                                 WihH, WhhH, WhrH, WfrH, bsum);

    rits_main<<<BB / 16, 512, 0, stream>>>(data, masks, deltas, Wtd, btd, bhr,
                                           bfr, WihH, WhhH, WhrH, WfrH, bsum,
                                           out, gNum, gDen);

    rits_finalize<<<1, SS, 0, stream>>>(gNum, gDen, lossOut);
}
